// SoftgroupAttention_16441134809375
// MI455X (gfx1250) — compile-verified, matrix operand order checked by reading, not run on hardware
//
#include <hip/hip_runtime.h>
#include <hip/hip_bf16.h>
#include <math.h>

// ---------------------------------------------------------------------------
// SoftgroupAttention on MI455X (gfx1250), wave32 + WMMA f16->f32.
// B=4, H=W=32 -> N=1024, C=384, 12 heads x hd=32, M=N/4=256 centers.
// ---------------------------------------------------------------------------

#define NB   4
#define NH   12
#define NN   1024
#define MM   256
#define CDIM 384
#define HD   32
#define BH   (NB * NH)   // 48

typedef __attribute__((ext_vector_type(16))) _Float16 v16h;
typedef __attribute__((ext_vector_type(8)))  _Float16 v8h;
typedef __attribute__((ext_vector_type(4)))  _Float16 v4h;
typedef __attribute__((ext_vector_type(8)))  float    v8f;
typedef __attribute__((ext_vector_type(4)))  float    v4f;

#define DEV __device__ __forceinline__

DEV v8f wmma16(v16h a, v16h b, v8f c) {
  // D = A(16x32 f16) * B(32x16 f16) + C(16x16 f32)
  return __builtin_amdgcn_wmma_f32_16x16x32_f16(
      /*neg_a=*/false, a, /*neg_b=*/false, b,
      /*c_mod=*/(short)0, c, /*reuse_a=*/false, /*reuse_b=*/false);
}

// Load a 16x32 f16 fragment from a row-major matrix (ld in halfs).
// 16-bit A layout: row = lane&15; lanes 0-15 hold K in {0..7, 16..23},
// lanes 16-31 hold K in {8..15, 24..31}.  K halves are contiguous in two
// 8-half (16-byte) chunks -> two b128 loads per lane.
// The B operand is loaded with the same pattern from the TRANSPOSED source
// (rows = output columns), i.e. pass Bt row-major.
DEV v16h load_frag(const _Float16* base, int ld, int lane) {
  int r = lane & 15;
  int g = (lane >> 4) & 1;
  const _Float16* row = base + (size_t)r * ld + 8 * g;
  v8h lo = *(const v8h*)(row);
  v8h hi = *(const v8h*)(row + 16);
  v16h a;
#pragma unroll
  for (int h = 0; h < 8; ++h) { a[h] = lo[h]; a[h + 8] = hi[h]; }
  return a;
}

DEV float sigmoidf_(float x) { return 1.0f / (1.0f + __expf(-x)); }

DEV v4h cvt4(v4f v) {
  v4h h;
#pragma unroll
  for (int i = 0; i < 4; ++i) h[i] = (_Float16)v[i];
  return h;
}

// ---------------------------------------------------------------------------
// K1: fused QKV + group projection.  x[4096,384] @ [Wqkv | Wg] (1536 cols).
// The 64-wide column tile never straddles the Wqkv/Wg boundary (1152%64==0),
// so the weight source is uniform per block: no divergence in staging.
// Staging is vectorized: b128 global loads -> f16 -> b64 LDS stores (A tile)
// or 4x b16 transposed stores (B tile).
// q is pre-scaled by hd^-0.5; k row-major; v stored transposed [bh][d][n];
// xg stored [b][n][384] fp32 (+bg) for the depthwise conv.
// ---------------------------------------------------------------------------
__global__ __launch_bounds__(128) void k_qkvg(
    const float* __restrict__ x, const float* __restrict__ Wqkv,
    const float* __restrict__ Wg, const float* __restrict__ bg,
    _Float16* __restrict__ qf16, _Float16* __restrict__ kf16,
    _Float16* __restrict__ vT, float* __restrict__ xg, float qscale) {
  __shared__ __align__(16) _Float16 As[16 * 40];
  __shared__ __align__(16) _Float16 Bs[64 * 40];
  int tid = threadIdx.x, lane = tid & 31, w = tid >> 5;
  int row0 = blockIdx.x * 16, c0 = blockIdx.y * 64;
  // Uniform weight source for this block's column tile.
  const float* Wbase;
  int ldw, cbase;
  if (c0 < 3 * CDIM) { Wbase = Wqkv; ldw = 3 * CDIM; cbase = c0; }
  else               { Wbase = Wg;   ldw = CDIM;     cbase = c0 - 3 * CDIM; }
  v8f acc = {};
  for (int k0 = 0; k0 < CDIM; k0 += 32) {
    __syncthreads();
    // A tile: 16x32 f32 = 512 elems, 4 per thread, one b128 load + b64 store.
    {
      int idx = tid * 4;                   // 0..508
      int r = idx >> 5, kk = idx & 31;
      v4f xv = *(const v4f*)(&x[(size_t)(row0 + r) * CDIM + k0 + kk]);
      *(v4h*)(&As[r * 40 + kk]) = cvt4(xv);
    }
    // B tile (transposed in LDS): 32x64 f32 = 2048 elems, 16 per thread.
#pragma unroll
    for (int it = 0; it < 4; ++it) {
      int idx = (tid + it * 128) * 4;      // 0..2044
      int kk = idx >> 6, c = idx & 63;
      v4f wv = *(const v4f*)(&Wbase[(size_t)(k0 + kk) * ldw + cbase + c]);
      v4h hv = cvt4(wv);
#pragma unroll
      for (int i = 0; i < 4; ++i) Bs[(c + i) * 40 + kk] = hv[i];
    }
    __syncthreads();
    v16h a = load_frag(As, 40, lane);
    v16h b = load_frag(Bs + w * 16 * 40, 40, lane);
    acc = wmma16(a, b, acc);
  }
  // C layout: col = lane&15; VGPR r -> row r + 8*(lane>>4)
  int g_ = (lane >> 4) & 1, cc = lane & 15;
  int cg = c0 + w * 16 + cc;
#pragma unroll
  for (int r = 0; r < 8; ++r) {
    int row = row0 + r + 8 * g_;
    int b_ = row >> 10, n = row & 1023;
    float v = acc[r];
    if (cg < 3 * CDIM) {
      int which = cg / CDIM, rem = cg % CDIM;
      int head = rem >> 5, d = rem & 31;
      int bh = b_ * NH + head;
      if (which == 0)       qf16[((size_t)bh * NN + n) * HD + d] = (_Float16)(v * qscale);
      else if (which == 1)  kf16[((size_t)bh * NN + n) * HD + d] = (_Float16)v;
      else                  vT[((size_t)bh * HD + d) * NN + n]   = (_Float16)v;
    } else {
      int c2 = cg - 3 * CDIM;
      xg[((size_t)b_ * NN + n) * CDIM + c2] = v + bg[c2];
    }
  }
}

// ---------------------------------------------------------------------------
// K2a: depthwise 3x3 stride-2 conv on xg -> group centers, then per-head
// l2norm.  Block = 384 thr (one channel each; each 32-channel head group is
// exactly one wave -> wave shuffle reduction).  gcn: [bh][m][32] f16.
// ---------------------------------------------------------------------------
__global__ __launch_bounds__(384) void k_conv_gcn(
    const float* __restrict__ xg, const float* __restrict__ Wconv,
    const float* __restrict__ bconv, _Float16* __restrict__ gcn) {
  int c = threadIdx.x;
  int b = blockIdx.y, yx = blockIdx.x;
  int y = yx >> 4, xp = yx & 15;
  float s = bconv[c];
#pragma unroll
  for (int dy = 0; dy < 3; ++dy)
#pragma unroll
    for (int dx = 0; dx < 3; ++dx) {
      int yy = 2 * y - 1 + dy, xx = 2 * xp - 1 + dx;
      if (yy >= 0 && yy < 32 && xx >= 0 && xx < 32)
        s += xg[(((size_t)b * 32 + yy) * 32 + xx) * CDIM + c] *
             Wconv[(dy * 3 + dx) * CDIM + c];
    }
  float t = s * s;
#pragma unroll
  for (int off = 16; off > 0; off >>= 1) t += __shfl_xor(t, off, 32);
  float inv = 1.0f / fmaxf(sqrtf(t), 1e-12f);
  int head = c >> 5, d = c & 31;
  gcn[(((size_t)b * NH + head) * MM + yx) * HD + d] = (_Float16)(s * inv);
}

// K2b: per-head l2norm of xg -> xgn [bh][n][32] f16.
__global__ __launch_bounds__(384) void k_xgn(
    const float* __restrict__ xg, _Float16* __restrict__ xgn) {
  int c = threadIdx.x;
  int row = blockIdx.x;           // b*1024 + n
  float s = xg[(size_t)row * CDIM + c];
  float t = s * s;
#pragma unroll
  for (int off = 16; off > 0; off >>= 1) t += __shfl_xor(t, off, 32);
  float inv = 1.0f / fmaxf(sqrtf(t), 1e-12f);
  int head = c >> 5, d = c & 31;
  int b = row >> 10, n = row & 1023;
  xgn[(((size_t)b * NH + head) * NN + n) * HD + d] = (_Float16)(s * inv);
}

// ---------------------------------------------------------------------------
// K3: sim = sigmoid(beta + alpha * xgn @ gcn^T)  [bh][1024][256] f16.
// One WMMA (K=32) per 16x16 tile; 4 waves cover 16 rows x 64 cols.
// ---------------------------------------------------------------------------
__global__ __launch_bounds__(128) void k_sim(
    const _Float16* __restrict__ xgn, const _Float16* __restrict__ gcn,
    const float* __restrict__ alpha, const float* __restrict__ beta,
    _Float16* __restrict__ simf) {
  int lane = threadIdx.x & 31, w = threadIdx.x >> 5;
  int bh = blockIdx.z;
  int n0 = blockIdx.x * 16;
  int m0 = blockIdx.y * 64 + w * 16;
  v16h a = load_frag(xgn + ((size_t)bh * NN + n0) * HD, HD, lane);
  v16h b = load_frag(gcn + ((size_t)bh * MM + m0) * HD, HD, lane);
  v8f acc = {};
  acc = wmma16(a, b, acc);
  float al = alpha[0], be = beta[0];
  int g_ = (lane >> 4) & 1, cc = lane & 15;
#pragma unroll
  for (int r = 0; r < 8; ++r) {
    int row = n0 + r + 8 * g_;
    simf[((size_t)bh * NN + row) * MM + m0 + cc] =
        (_Float16)sigmoidf_(be + al * acc[r]);
  }
}

// ---------------------------------------------------------------------------
// K3b: row softmax stats of S = (q*scale) @ k^T.  One row per thread,
// k tiles staged in LDS (read back as b128 vectors), online max/sum.
// ---------------------------------------------------------------------------
__global__ __launch_bounds__(128) void k_rowstats(
    const _Float16* __restrict__ qf16, const _Float16* __restrict__ kf16,
    float* __restrict__ rmax, float* __restrict__ rsum) {
  __shared__ __align__(16) _Float16 kt[32 * HD];
  int tid = threadIdx.x;
  int bh = blockIdx.y;
  int n = blockIdx.x * 128 + tid;
  float q[HD];
  const _Float16* qrow = qf16 + ((size_t)bh * NN + n) * HD;
#pragma unroll
  for (int j = 0; j < HD; ++j) q[j] = (float)qrow[j];
  float mx = -3.0e38f, sm = 0.0f;
  for (int m0 = 0; m0 < NN; m0 += 32) {
    __syncthreads();
    ((v8h*)kt)[tid] = ((const v8h*)(kf16 + ((size_t)bh * NN + m0) * HD))[tid];
    __syncthreads();
#pragma unroll 4
    for (int mm = 0; mm < 32; ++mm) {
      const v8h* kr = (const v8h*)(kt + mm * HD);
      float s = 0.0f;
#pragma unroll
      for (int ch = 0; ch < 4; ++ch) {
        v8h k8 = kr[ch];
#pragma unroll
        for (int j = 0; j < 8; ++j) s += q[ch * 8 + j] * (float)k8[j];
      }
      float nm = fmaxf(mx, s);
      sm = sm * __expf(mx - nm) + __expf(s - nm);
      mx = nm;
    }
  }
  rmax[(size_t)bh * NN + n] = mx;
  rsum[(size_t)bh * NN + n] = sm;
}

// ---------------------------------------------------------------------------
// K4: column sums of attn = softmax(S) * (gamma*sim@sim^T + 1-gamma).
// Wave owns 16 columns m; loops all 1024 rows in 16-row WMMA tiles,
// recomputing S (1 wmma) and gw (8 wmmas, K=256) per tile.
// ---------------------------------------------------------------------------
__global__ __launch_bounds__(128) void k_colsum(
    const _Float16* __restrict__ qf16, const _Float16* __restrict__ kf16,
    const _Float16* __restrict__ simf, const float* __restrict__ rmax,
    const float* __restrict__ rsum, const float* __restrict__ ggamma,
    float* __restrict__ colsum) {
  int lane = threadIdx.x & 31, w = threadIdx.x >> 5;
  int g_ = (lane >> 4) & 1;
  int bh = blockIdx.y;
  int m0 = blockIdx.x * 64 + w * 16;
  float gamma = sigmoidf_(ggamma[0]);
  // B-side fragments are fixed for this wave's columns: hoist.
  v16h kb = load_frag(kf16 + ((size_t)bh * NN + m0) * HD, HD, lane);
  v16h sb[8];
#pragma unroll
  for (int j = 0; j < 8; ++j)
    sb[j] = load_frag(simf + ((size_t)bh * NN + m0) * MM + j * 32, MM, lane);
  float csum = 0.0f;
  for (int n0 = 0; n0 < NN; n0 += 16) {
    // prefetch next n-tile of sim (the dominant 25MB stream)
    if (n0 + 16 < NN)
      __builtin_prefetch(simf + ((size_t)bh * NN + n0 + 16) * MM, 0, 3);
    v16h qa = load_frag(qf16 + ((size_t)bh * NN + n0) * HD, HD, lane);
    v8f sacc = {};
    sacc = wmma16(qa, kb, sacc);
    v8f gacc = {};
#pragma unroll
    for (int j = 0; j < 8; ++j) {
      v16h sa = load_frag(simf + ((size_t)bh * NN + n0) * MM + j * 32, MM, lane);
      gacc = wmma16(sa, sb[j], gacc);
    }
#pragma unroll
    for (int r = 0; r < 8; ++r) {
      int row = n0 + r + 8 * g_;
      float rm = rmax[(size_t)bh * NN + row];
      float rs = rsum[(size_t)bh * NN + row];
      csum += __expf(sacc[r] - rm) / rs * (gamma * gacc[r] + 1.0f - gamma);
    }
  }
  // lanes L and L+16 hold the same column (different row halves)
  csum += __shfl_xor(csum, 16, 32);
  if (lane < 16) colsum[(size_t)bh * NN + m0 + lane] = csum;
}

// K4b: fold 1/(colsum+1e-8) into transposed v.
__global__ __launch_bounds__(256) void k_scalev(
    const _Float16* __restrict__ vT, const float* __restrict__ colsum,
    _Float16* __restrict__ vT2) {
  int idx = blockIdx.x * 256 + threadIdx.x;   // [bh][d][m], 48*32*1024
  int m = idx & 1023;
  int bh = idx >> 15;                          // idx / (32*1024)
  float inv = 1.0f / (colsum[(size_t)bh * NN + m] + 1e-8f);
  vT2[idx] = (_Float16)((float)vT[idx] * inv);
}

// ---------------------------------------------------------------------------
// K5: out = attn_norm @ v'.  Wave owns 16 rows x 16 of the 32 d-columns.
// Per 32-wide m step: recompute attn tile (S + gw wmmas), stage as f16 in a
// wave-private LDS tile (s_wait_dscnt for wave-local coherence), reload as an
// A fragment, and wmma against v'^T.  Result scattered head-major into xo.
// ---------------------------------------------------------------------------
__global__ __launch_bounds__(128) void k_out(
    const _Float16* __restrict__ qf16, const _Float16* __restrict__ kf16,
    const _Float16* __restrict__ simf, const _Float16* __restrict__ vT2,
    const float* __restrict__ rmax, const float* __restrict__ rsum,
    const float* __restrict__ ggamma, float* __restrict__ xo) {
  __shared__ __align__(16) _Float16 lattn[4][16 * 40];
  int lane = threadIdx.x & 31, w = threadIdx.x >> 5;
  int g_ = (lane >> 4) & 1;
  int bh = blockIdx.y;
  int n0 = blockIdx.x * 32 + (w >> 1) * 16;
  int d0 = (w & 1) * 16;
  float gamma = sigmoidf_(ggamma[0]);
  float rmx[8], rsm[8];
#pragma unroll
  for (int r = 0; r < 8; ++r) {
    int row = n0 + r + 8 * g_;
    rmx[r] = rmax[(size_t)bh * NN + row];
    rsm[r] = rsum[(size_t)bh * NN + row];
  }
  v16h qa = load_frag(qf16 + ((size_t)bh * NN + n0) * HD, HD, lane);
  _Float16* lat = &lattn[w][0];
  v8f oacc = {};
  for (int m0 = 0; m0 < NN; m0 += 32) {
    if (m0 + 32 < NN)
      __builtin_prefetch(simf + ((size_t)bh * NN + m0 + 32) * MM, 0, 3);
#pragma unroll
    for (int mt = 0; mt < 2; ++mt) {
      int mb = m0 + mt * 16;
      v16h kb = load_frag(kf16 + ((size_t)bh * NN + mb) * HD, HD, lane);
      v8f sacc = {};
      sacc = wmma16(qa, kb, sacc);
      v8f gacc = {};
#pragma unroll
      for (int j = 0; j < 8; ++j) {
        v16h sa = load_frag(simf + ((size_t)bh * NN + n0) * MM + j * 32, MM, lane);
        v16h sbj = load_frag(simf + ((size_t)bh * NN + mb) * MM + j * 32, MM, lane);
        gacc = wmma16(sa, sbj, gacc);
      }
      int col = (lane & 15) + mt * 16;
#pragma unroll
      for (int r = 0; r < 8; ++r) {
        float a = __expf(sacc[r] - rmx[r]) / rsm[r] *
                  (gamma * gacc[r] + 1.0f - gamma);
        lat[(r + 8 * g_) * 40 + col] = (_Float16)a;
      }
    }
    // Wave-private LDS tile: only this wave's DS ops need to land.
    asm volatile("s_wait_dscnt 0" ::: "memory");
    v16h aat = load_frag(lat, 40, lane);
    v16h vb = load_frag(vT2 + ((size_t)bh * HD + d0) * NN + m0, NN, lane);
    oacc = wmma16(aat, vb, oacc);
  }
  int b = bh / NH, head = bh % NH;
  int c = head * HD + d0 + (lane & 15);
#pragma unroll
  for (int r = 0; r < 8; ++r) {
    int n = n0 + r + 8 * g_;
    xo[((size_t)b * NN + n) * CDIM + c] = oacc[r];
  }
}

// ---------------------------------------------------------------------------
// K6: final projection xo[4096,384] @ Wproj[384,384] -> out.
// Same vectorized staging as K1.
// ---------------------------------------------------------------------------
__global__ __launch_bounds__(128) void k_proj(
    const float* __restrict__ xo, const float* __restrict__ Wp,
    float* __restrict__ out) {
  __shared__ __align__(16) _Float16 As[16 * 40];
  __shared__ __align__(16) _Float16 Bs[64 * 40];
  int tid = threadIdx.x, lane = tid & 31, w = tid >> 5;
  int row0 = blockIdx.x * 16, c0 = blockIdx.y * 64;
  v8f acc = {};
  for (int k0 = 0; k0 < CDIM; k0 += 32) {
    __syncthreads();
    {
      int idx = tid * 4;
      int r = idx >> 5, kk = idx & 31;
      v4f xv = *(const v4f*)(&xo[(size_t)(row0 + r) * CDIM + k0 + kk]);
      *(v4h*)(&As[r * 40 + kk]) = cvt4(xv);
    }
#pragma unroll
    for (int it = 0; it < 4; ++it) {
      int idx = (tid + it * 128) * 4;
      int kk = idx >> 6, c = idx & 63;
      v4f wv = *(const v4f*)(&Wp[(size_t)(k0 + kk) * CDIM + c0 + c]);
      v4h hv = cvt4(wv);
#pragma unroll
      for (int i = 0; i < 4; ++i) Bs[(c + i) * 40 + kk] = hv[i];
    }
    __syncthreads();
    v16h a = load_frag(As, 40, lane);
    v16h b = load_frag(Bs + w * 16 * 40, 40, lane);
    acc = wmma16(a, b, acc);
  }
  int g_ = (lane >> 4) & 1, cc = lane & 15;
#pragma unroll
  for (int r = 0; r < 8; ++r) {
    int row = row0 + r + 8 * g_;
    out[(size_t)row * CDIM + c0 + w * 16 + cc] = acc[r];
  }
}

// ---------------------------------------------------------------------------
// Host orchestration.
// ---------------------------------------------------------------------------
extern "C" void kernel_launch(void* const* d_in, const int* in_sizes, int n_in,
                              void* d_out, int out_size, void* d_ws,
                              size_t ws_size, hipStream_t stream) {
  (void)in_sizes; (void)n_in; (void)out_size; (void)ws_size;
  const float* x           = (const float*)d_in[0];
  const float* Wqkv        = (const float*)d_in[1];
  const float* Wg          = (const float*)d_in[2];
  const float* bg          = (const float*)d_in[3];
  const float* Wconv       = (const float*)d_in[4];
  const float* bconv       = (const float*)d_in[5];
  const float* Wproj       = (const float*)d_in[6];
  const float* sim_alpha   = (const float*)d_in[7];
  const float* sim_beta    = (const float*)d_in[8];
  const float* group_gamma = (const float*)d_in[9];
  float* out = (float*)d_out;

  char* p = (char*)d_ws;
  _Float16* qf16 = (_Float16*)(p + 0);          // 48*1024*32 f16 = 3 MB
  _Float16* kf16 = (_Float16*)(p + 3145728);
  _Float16* vT   = (_Float16*)(p + 6291456);    // [bh][d][n]
  _Float16* vT2  = (_Float16*)(p + 9437184);    // v^T scaled by 1/colsum
  float*    xg   = (float*)   (p + 12582912);   // [b][n][384] fp32, 6 MB
  _Float16* xgn  = (_Float16*)(p + 18874368);   // [bh][n][32]
  _Float16* gcn  = (_Float16*)(p + 22020096);   // [bh][m][32]
  _Float16* simf = (_Float16*)(p + 22806528);   // [bh][1024][256] f16, 25 MB
  float*    rmax = (float*)   (p + 47972352);   // [bh][1024]
  float*    rsum = (float*)   (p + 48168960);
  float*    csum = (float*)   (p + 48365568);
  float*    xo   = (float*)   (p + 48562176);   // [4096][384] fp32
  // total workspace footprint: 54,853,632 bytes (~52 MB) -> L2 resident.

  const float qscale = 0.17677669529663688f;    // 32^-0.5

  k_qkvg<<<dim3(NB * NN / 16, (3 * CDIM + CDIM) / 64), 128, 0, stream>>>(
      x, Wqkv, Wg, bg, qf16, kf16, vT, xg, qscale);
  k_conv_gcn<<<dim3(MM, NB), 384, 0, stream>>>(xg, Wconv, bconv, gcn);
  k_xgn<<<dim3(NB * NN), 384, 0, stream>>>(xg, xgn);
  k_sim<<<dim3(NN / 16, MM / 64, BH), 128, 0, stream>>>(
      xgn, gcn, sim_alpha, sim_beta, simf);
  k_rowstats<<<dim3(NN / 128, BH), 128, 0, stream>>>(qf16, kf16, rmax, rsum);
  k_colsum<<<dim3(NN / 64, BH), 128, 0, stream>>>(
      qf16, kf16, simf, rmax, rsum, group_gamma, csum);
  k_scalev<<<dim3(BH * HD * NN / 256), 256, 0, stream>>>(vT, csum, vT2);
  k_out<<<dim3(NN / 32, BH), 128, 0, stream>>>(
      qf16, kf16, simf, vT2, rmax, rsum, group_gamma, xo);
  k_proj<<<dim3(NB * NN / 16, CDIM / 64), 128, 0, stream>>>(xo, Wproj, out);
}
